// NLB_6717328850962
// MI455X (gfx1250) — matrix operand order repaired and checked by reading, not compiled, not run on hardware
//
#include <hip/hip_runtime.h>

typedef _Float16 f16;
typedef __attribute__((ext_vector_type(16))) _Float16 v16h;
typedef __attribute__((ext_vector_type(8)))  _Float16 v8h;
typedef __attribute__((ext_vector_type(8)))  float    v8f;
typedef __attribute__((ext_vector_type(4)))  int      v4i;

union V16 { v16h v; v8h h[2]; };

#define BATCH 8
#define CCH   128
#define INTER 64
#define NPIX  4096   // 64*64
#define KT    64     // keys per tile
#define NT    (NPIX / KT)

// ---- async global->LDS (CDNA5) with compile-time fallback ------------------
#if defined(__has_builtin)
#  if __has_builtin(__builtin_amdgcn_global_load_async_to_lds_b128)
#    define USE_ASYNC 1
#  endif
#endif
#ifndef USE_ASYNC
#  define USE_ASYNC 0
#endif

typedef __attribute__((address_space(1))) v4i* gv4_t;   // global int4*
typedef __attribute__((address_space(3))) v4i* lv4_t;   // LDS int4*

__device__ __forceinline__ void async_wait0() {
#if USE_ASYNC
#  if __has_builtin(__builtin_amdgcn_s_wait_asynccnt)
    __builtin_amdgcn_s_wait_asynccnt(0);
#  else
    asm volatile("s_wait_asynccnt 0" ::: "memory");
#  endif
#endif
}

// ---- DPP 16-lane row reductions (stay inside each wave32 half) -------------
// xor1 -> xor2 -> xor7(row_half_mirror) -> xor15(row_mirror) = full 16-lane tree
#define DPPF(x, ctrl)                                                          \
    __int_as_float(__builtin_amdgcn_mov_dpp(__float_as_int(x), (ctrl), 0xf, 0xf, true))

__device__ __forceinline__ float row_max16(float m) {
    m = fmaxf(m, DPPF(m, 0xB1));   // quad_perm [1,0,3,2]  (xor 1)
    m = fmaxf(m, DPPF(m, 0x4E));   // quad_perm [2,3,0,1]  (xor 2)
    m = fmaxf(m, DPPF(m, 0x141));  // row_half_mirror      (xor 7)
    m = fmaxf(m, DPPF(m, 0x140));  // row_mirror           (xor 15)
    return m;
}
__device__ __forceinline__ float row_sum16(float s) {
    s += DPPF(s, 0xB1);
    s += DPPF(s, 0x4E);
    s += DPPF(s, 0x141);
    s += DPPF(s, 0x140);
    return s;
}

__device__ __forceinline__ v8f wmma_f16(v16h a, v16h b, v8f c) {
    return __builtin_amdgcn_wmma_f32_16x16x32_f16(
        false, a, false, b, (short)0, c, false, false);
}

// ---------------------------------------------------------------------------
// Phase 1: 1x1 conv projections. theta/phiT stored [b, n, 64] f16; g stored
// transposed [b, 64, n] f16 so phase-2 B fragments are contiguous loads.
// ---------------------------------------------------------------------------
__global__ __launch_bounds__(256) void nlb_proj(
    const float* __restrict__ x,
    const float* __restrict__ tw, const float* __restrict__ tb,
    const float* __restrict__ pw, const float* __restrict__ pb,
    const float* __restrict__ gw, const float* __restrict__ gb,
    f16* __restrict__ theta, f16* __restrict__ phiT, f16* __restrict__ gT)
{
    __shared__ float xs[CCH][64];
    const int b  = blockIdx.y;
    const int n0 = blockIdx.x * 64;
    const int tid = threadIdx.x;

    for (int idx = tid; idx < CCH * 64; idx += 256) {
        const int c = idx >> 6, j = idx & 63;
        xs[c][j] = x[((size_t)b * CCH + c) * NPIX + n0 + j];
    }
    __syncthreads();

    const int j   = tid & 63;
    const int grp = tid >> 6;

    for (int oi = 0; oi < 16; ++oi) {
        const int o = grp * 16 + oi;
        float at = tb[o], ap = pb[o], ag = gb[o];
        const float* twr = tw + o * CCH;
        const float* pwr = pw + o * CCH;
        const float* gwr = gw + o * CCH;
        #pragma unroll 8
        for (int c = 0; c < CCH; ++c) {
            const float xv = xs[c][j];
            at = fmaf(twr[c], xv, at);
            ap = fmaf(pwr[c], xv, ap);
            ag = fmaf(gwr[c], xv, ag);
        }
        theta[((size_t)b * NPIX + n0 + j) * INTER + o] = (f16)at;
        phiT [((size_t)b * NPIX + n0 + j) * INTER + o] = (f16)ap;
        gT   [((size_t)b * INTER + o) * NPIX + n0 + j] = (f16)ag;
    }
}

// ---------------------------------------------------------------------------
// Phase 2: flash attention. 4 waves/WG, one 16-query strip per wave.
// phi/g tiles (64 keys) staged in LDS, shared by all 4 waves, double-buffered
// via async global->LDS DMA so tile t+1 streams in while tile t computes.
// ---------------------------------------------------------------------------
__device__ __forceinline__ void stage_tile(
    const f16* __restrict__ phiT, const f16* __restrict__ gT,
    f16* sphi, f16* sg, int b, int j0, int wid, int lane)
{
    // phi tile: [64 keys][64 k] = 8KB, globally contiguous
    const char* psrc = (const char*)(phiT + ((size_t)b * NPIX + j0) * INTER);
#if USE_ASYNC
    #pragma unroll
    for (int i = 0; i < 4; ++i) {
        const int off = (wid * 4 + i) * 512 + lane * 16;
        __builtin_amdgcn_global_load_async_to_lds_b128(
            (gv4_t)(psrc + off), (lv4_t)((char*)sphi + off), 0, 0);
    }
    // g tile: 64 inter rows x 128B, global row stride 8KB
    const char* gsrc = (const char*)gT;
    #pragma unroll
    for (int i = 0; i < 4; ++i) {
        const int r0 = (wid * 4 + i) * 4 + (lane >> 3);
        const size_t go = ((size_t)(b * INTER + r0) * NPIX + j0) * 2 + (lane & 7) * 16;
        const int    lo = r0 * 128 + (lane & 7) * 16;
        __builtin_amdgcn_global_load_async_to_lds_b128(
            (gv4_t)(gsrc + go), (lv4_t)((char*)sg + lo), 0, 0);
    }
#else
    const int tid = wid * 32 + lane;
    #pragma unroll
    for (int i = 0; i < 4; ++i) {
        const int h = i * 1024 + tid * 8;            // half-index 0..4095
        *(v8h*)((char*)sphi + h * 2) = *(const v8h*)(psrc + h * 2);
        const int row = h >> 6, col = h & 63;
        *(v8h*)(sg + row * 64 + col) =
            *(const v8h*)(gT + (size_t)(b * INTER + row) * NPIX + j0 + col);
    }
#endif
}

__global__ __launch_bounds__(128) void nlb_attn(
    const f16* __restrict__ theta, const f16* __restrict__ phiT,
    const f16* __restrict__ gT, float* __restrict__ y)
{
    __shared__ __align__(16) f16 sphi[2][KT * INTER];     // 2 x 8KB
    __shared__ __align__(16) f16 sg  [2][INTER * KT];     // 2 x 8KB
    __shared__ __align__(16) f16 pstage[4][16 * KT];      // 4 x 2KB

    const int b    = blockIdx.y;
    const int lane = threadIdx.x & 31;
    const int wid  = threadIdx.x >> 5;
    const int q0   = blockIdx.x * 64 + wid * 16;
    const int ml   = lane & 15;
    const int hi   = lane >> 4;

    // theta A fragments (16q x 64k): elems 0-7 K=ka*32+hi*8+e, 8-15 K=+16
    const f16* th = theta + ((size_t)b * NPIX + q0 + ml) * INTER;
    V16 a0, a1;
    a0.h[0] = *(const v8h*)(th      + hi * 8);
    a0.h[1] = *(const v8h*)(th + 16 + hi * 8);
    a1.h[0] = *(const v8h*)(th + 32      + hi * 8);
    a1.h[1] = *(const v8h*)(th + 32 + 16 + hi * 8);

    v8f of[4] = {};
    float mrun[8], lrun[8];
    #pragma unroll
    for (int r = 0; r < 8; ++r) { mrun[r] = -1e30f; lrun[r] = 0.0f; }

    f16* ps = &pstage[wid][0];

    stage_tile(phiT, gT, sphi[0], sg[0], b, 0, wid, lane);

    for (int t = 0; t < NT; ++t) {
        async_wait0();          // this wave's portion of tile t has landed
        __syncthreads();        // everyone's portion landed; tile t-1 consumed
        if (t + 1 < NT)
            stage_tile(phiT, gT, sphi[(t + 1) & 1], sg[(t + 1) & 1],
                       b, (t + 1) * KT, wid, lane);

        const f16* cphi = &sphi[t & 1][0];
        const f16* cg   = &sg  [t & 1][0];

        // --- S = theta * phi^T for 64 keys (4 x 16-key column tiles) -------
        v8f s[4];
        #pragma unroll
        for (int kk = 0; kk < 4; ++kk) {
            const f16* prow = cphi + (kk * 16 + ml) * INTER + hi * 8;
            V16 b0, b1;
            b0.h[0] = *(const v8h*)(prow);      b0.h[1] = *(const v8h*)(prow + 16);
            b1.h[0] = *(const v8h*)(prow + 32); b1.h[1] = *(const v8h*)(prow + 48);
            v8f acc = {};
            acc = wmma_f16(a0.v, b0.v, acc);
            acc = wmma_f16(a1.v, b1.v, acc);
            s[kk] = acc;
        }

        // --- online softmax: row q = r + 8*hi lives in one 16-lane half ----
        float alpha[8];
        #pragma unroll
        for (int r = 0; r < 8; ++r) {
            float mr = fmaxf(fmaxf(s[0][r], s[1][r]), fmaxf(s[2][r], s[3][r]));
            mr = row_max16(mr);
            const float mnew = fmaxf(mrun[r], mr);
            const float al   = __expf(mrun[r] - mnew);
            const float e0 = __expf(s[0][r] - mnew);
            const float e1 = __expf(s[1][r] - mnew);
            const float e2 = __expf(s[2][r] - mnew);
            const float e3 = __expf(s[3][r] - mnew);
            const float rs = row_sum16((e0 + e1) + (e2 + e3));
            lrun[r]  = lrun[r] * al + rs;
            mrun[r]  = mnew;
            alpha[r] = al;
            f16* pr = ps + (r + 8 * hi) * KT + ml;
            pr[0]  = (f16)e0;
            pr[16] = (f16)e1;
            pr[32] = (f16)e2;
            pr[48] = (f16)e3;
        }
        #pragma unroll
        for (int nt = 0; nt < 4; ++nt) {
            #pragma unroll
            for (int r = 0; r < 8; ++r) of[nt][r] *= alpha[r];
        }

        // --- reload P as two 16x32 A fragments (ds_load_b128) --------------
        V16 pa0, pa1;
        {
            const f16* pr = ps + ml * KT;
            pa0.h[0] = *(const v8h*)(pr      + hi * 8);
            pa0.h[1] = *(const v8h*)(pr + 16 + hi * 8);
            pa1.h[0] = *(const v8h*)(pr + 32      + hi * 8);
            pa1.h[1] = *(const v8h*)(pr + 32 + 16 + hi * 8);
        }

        // --- O += P * g --------------------------------------------------
        #pragma unroll
        for (int nt = 0; nt < 4; ++nt) {
            const f16* grow = cg + (nt * 16 + ml) * KT + hi * 8;
            V16 g0, g1;
            g0.h[0] = *(const v8h*)(grow);      g0.h[1] = *(const v8h*)(grow + 16);
            g1.h[0] = *(const v8h*)(grow + 32); g1.h[1] = *(const v8h*)(grow + 48);
            of[nt] = wmma_f16(pa0.v, g0.v, of[nt]);
            of[nt] = wmma_f16(pa1.v, g1.v, of[nt]);
        }
    }

    // --- normalize, write y as [b, n, 64] f32 ------------------------------
    float* yb = y + ((size_t)b * NPIX + q0) * INTER;
    #pragma unroll
    for (int r = 0; r < 8; ++r) {
        const float inv = 1.0f / lrun[r];
        #pragma unroll
        for (int nt = 0; nt < 4; ++nt)
            yb[(r + 8 * hi) * INTER + nt * 16 + ml] = of[nt][r] * inv;
    }
}

// ---------------------------------------------------------------------------
// Phase 3: out = out_w @ y + out_b + x (residual)
// ---------------------------------------------------------------------------
__global__ __launch_bounds__(256) void nlb_out(
    const float* __restrict__ x,  const float* __restrict__ ow,
    const float* __restrict__ ob, const float* __restrict__ y,
    float* __restrict__ out)
{
    __shared__ float wsh[CCH * INTER];
    __shared__ float ysh[64 * (INTER + 1)];
    const int b  = blockIdx.y;
    const int n0 = blockIdx.x * 64;
    const int tid = threadIdx.x;

    for (int idx = tid; idx < CCH * INTER; idx += 256) wsh[idx] = ow[idx];
    for (int idx = tid; idx < 64 * INTER; idx += 256) {
        const int j = idx >> 6, o = idx & 63;
        ysh[j * (INTER + 1) + o] = y[((size_t)b * NPIX + n0 + j) * INTER + o];
    }
    __syncthreads();

    for (int idx = tid; idx < CCH * 64; idx += 256) {
        const int c = idx >> 6, j = idx & 63;
        float acc = ob[c];
        const float* wr = wsh + c * INTER;
        const float* yr = ysh + j * (INTER + 1);
        #pragma unroll 8
        for (int o = 0; o < INTER; ++o) acc = fmaf(wr[o], yr[o], acc);
        const size_t gi = ((size_t)b * CCH + c) * NPIX + n0 + j;
        out[gi] = acc + x[gi];
    }
}

// ---------------------------------------------------------------------------
extern "C" void kernel_launch(void* const* d_in, const int* in_sizes, int n_in,
                              void* d_out, int out_size, void* d_ws, size_t ws_size,
                              hipStream_t stream) {
    const float* x  = (const float*)d_in[0];
    const float* tw = (const float*)d_in[1];
    const float* tb = (const float*)d_in[2];
    const float* pw = (const float*)d_in[3];
    const float* pb = (const float*)d_in[4];
    const float* gw = (const float*)d_in[5];
    const float* gb = (const float*)d_in[6];
    const float* ow = (const float*)d_in[7];
    const float* ob = (const float*)d_in[8];
    float* out = (float*)d_out;

    char* ws = (char*)d_ws;
    f16*   theta = (f16*)(ws);                        // 4 MB
    f16*   phiT  = (f16*)(ws + ((size_t)4  << 20));   // 4 MB
    f16*   gT    = (f16*)(ws + ((size_t)8  << 20));   // 4 MB
    float* y     = (float*)(ws + ((size_t)12 << 20)); // 8 MB

    dim3 grd(NPIX / 64, BATCH);
    nlb_proj<<<grd, dim3(256), 0, stream>>>(x, tw, tb, pw, pb, gw, gb,
                                            theta, phiT, gT);
    nlb_attn<<<grd, dim3(128), 0, stream>>>(theta, phiT, gT, y);
    nlb_out <<<grd, dim3(256), 0, stream>>>(x, ow, ob, y, out);
}
